// Luong_full_90589450207862
// MI455X (gfx1250) — matrix operand order repaired and checked by reading, not compiled, not run on hardware
//
#include <hip/hip_runtime.h>

// ---------------------------------------------------------------------------
// Luong seq2seq (biGRU encoder + attention + GRU decoder) for gfx1250.
// All matmuls run through one bf16 WMMA GEMM (v_wmma_f32_16x16x32_bf16).
// Each wave32 owns a 32x64 output tile: 2 A-frags x 4 B-frags = 8 WMMAs per
// 32-wide K step, so every loaded B-fragment (weight data) feeds 2 WMMAs.
// Sequential GRU recurrences are per-step GEMM+gate launches serialized by
// the stream; everything else (input projections, attention, joiner, 32k
// vocab projection) is batched into large WMMA GEMMs.
// ---------------------------------------------------------------------------

typedef unsigned short u16;
typedef u16   u16x8 __attribute__((ext_vector_type(8)));
typedef __bf16 bf16;
typedef bf16  v16bf __attribute__((ext_vector_type(16)));
typedef float v8f   __attribute__((ext_vector_type(8)));

union FragCvt { v16bf v; u16x8 h[2]; };

__device__ __forceinline__ u16 f2bf(float f) {
  union { float f; unsigned u; } cv; cv.f = f;
  unsigned u = cv.u;
  u += 0x7fffu + ((u >> 16) & 1u);          // round-to-nearest-even
  return (u16)(u >> 16);
}

__device__ __forceinline__ float sigm(float x) { return 1.0f / (1.0f + __expf(-x)); }

// ---------------------------------------------------------------------------
// Generic GEMM: C[M,N] = A[M,K] * W[N,K]^T (+bias) with optional tanh.
// A, W are bf16 (stored as u16). Per-grid.z batch with independent strides.
// Block = 32 threads = 1 wave; wave computes 32 rows x 64 cols.
// Fragment layouts per CDNA5 ISA 7.12.2:
//   A 16x32: lane (half=l>>4, m=l&15), elems e0..7 -> K=8*half+e,
//            e8..15 -> K=16+8*half+(e-8)   (two contiguous 8xK chunks)
//   B 32x16: lane n=l&15, elems e0..15 -> same K as A (two contiguous 8xK chunks)
//   C 16x16: VGPR r, lanes0-15 -> M=r, lanes16-31 -> M=8+r; N=lane&15
// ---------------------------------------------------------------------------
__global__ void k_gemm(const u16* __restrict__ A, long lda, long bsA,
                       const u16* __restrict__ W, long ldw, long bsW,
                       float* __restrict__ C, long ldc, long bsC,
                       const float* __restrict__ bias, long bsBias,
                       int K, int act)
{
  const int lane = threadIdx.x & 31;
  const int l15  = lane & 15;
  const int half = lane >> 4;
  const long batch = blockIdx.z;
  A += batch * bsA;  W += batch * bsW;  C += batch * bsC;
  if (bias) bias += batch * bsBias;

  const long m0 = (long)blockIdx.y * 32;
  const long n0 = (long)blockIdx.x * 64;

  v8f acc[2][4] = {};
  const u16* arow0 = A + (m0 + l15) * lda + 8 * half;
  const u16* arow1 = A + (m0 + 16 + l15) * lda + 8 * half;
  const u16* wrow[4];
#pragma unroll
  for (int f = 0; f < 4; ++f)
    wrow[f] = W + (n0 + f * 16 + l15) * ldw + 8 * half;

  for (int k0 = 0; k0 < K; k0 += 32) {
    FragCvt a0, a1;
    a0.h[0] = *(const u16x8*)(arow0 + k0);
    a0.h[1] = *(const u16x8*)(arow0 + k0 + 16);
    a1.h[0] = *(const u16x8*)(arow1 + k0);
    a1.h[1] = *(const u16x8*)(arow1 + k0 + 16);
#pragma unroll
    for (int f = 0; f < 4; ++f) {
      FragCvt bw;
      bw.h[0] = *(const u16x8*)(wrow[f] + k0);
      bw.h[1] = *(const u16x8*)(wrow[f] + k0 + 16);
      acc[0][f] = __builtin_amdgcn_wmma_f32_16x16x32_bf16(
          false, a0.v, false, bw.v, (short)0, acc[0][f], false, false);
      acc[1][f] = __builtin_amdgcn_wmma_f32_16x16x32_bf16(
          false, a1.v, false, bw.v, (short)0, acc[1][f], false, false);
    }
  }

#pragma unroll
  for (int p = 0; p < 2; ++p) {
#pragma unroll
    for (int f = 0; f < 4; ++f) {
      long n = n0 + f * 16 + l15;
      float bv = bias ? bias[n] : 0.0f;
#pragma unroll
      for (int r = 0; r < 8; ++r) {
        float v = acc[p][f][r] + bv;
        if (act) v = tanhf(v);
        C[(m0 + p * 16 + 8 * half + r) * ldc + n] = v;
      }
    }
  }
}

// --------------------------- small support kernels -------------------------

__global__ void k_cvt(const float* __restrict__ in, u16* __restrict__ out, long n) {
  long i = (long)blockIdx.x * 256 + threadIdx.x;
  if (i < n) out[i] = f2bf(in[i]);
}

__global__ void k_zero(unsigned* __restrict__ p, long n) {
  long i = (long)blockIdx.x * 256 + threadIdx.x;
  if (i < n) p[i] = 0u;
}

// gather encoder embeddings -> bf16 [L*B, 1024]
__global__ void k_embed_enc(const int* __restrict__ tok,
                            const float* __restrict__ emb,
                            u16* __restrict__ out) {
  long row = blockIdx.x;                       // t*B + b
  long tk  = tok[row];
  const float* e = emb + tk * 1024;
  u16* o = out + row * 1024;
  for (int j = threadIdx.x; j < 1024; j += blockDim.x) o[j] = f2bf(e[j]);
}

// decoder teacher-forcing inputs: tok_in[0]=SOS(1), tok_in[t]=output_tokens[t-1]
__global__ void k_embed_dec(const int* __restrict__ out_tokens,
                            const float* __restrict__ emb,
                            u16* __restrict__ out) {
  long row = blockIdx.x;                       // t*32 + b
  int t = (int)(row >> 5), b = (int)(row & 31);
  long tk = (t == 0) ? 1 : out_tokens[(long)(t - 1) * 32 + b];
  const float* e = emb + tk * 1024;
  u16* o = out + row * 1024;
  for (int j = threadIdx.x; j < 1024; j += blockDim.x) o[j] = f2bf(e[j]);
}

// encoder GRU gate update (both directions via blockIdx.y); HH=512, L=128
__global__ void k_gru_gates_enc(const float* __restrict__ gi,   // [2][L*32][1536]
                                const float* __restrict__ gh,   // [2][32][1536]
                                float* __restrict__ h_f,        // [2][32][512]
                                u16*   __restrict__ h_bf,
                                float* __restrict__ y,          // [L][32][1024]
                                int t) {
  int dir = blockIdx.y;
  int idx = blockIdx.x * 256 + threadIdx.x;    // 0 .. 32*512-1
  int b = idx >> 9, j = idx & 511;
  int td = dir ? (127 - t) : t;
  const float* gir = gi + (long)dir * 128 * 32 * 1536 + ((long)td * 32 + b) * 1536;
  const float* ghr = gh + (long)dir * 32 * 1536 + (long)b * 1536;
  float r = sigm(gir[j] + ghr[j]);
  float z = sigm(gir[512 + j] + ghr[512 + j]);
  float n = tanhf(gir[1024 + j] + r * ghr[1024 + j]);
  long hoff = (long)dir * 32 * 512 + (long)b * 512 + j;
  float hn = (1.0f - z) * n + z * h_f[hoff];
  h_f[hoff] = hn;  h_bf[hoff] = f2bf(hn);
  y[((long)td * 32 + b) * 1024 + (long)dir * 512 + j] = hn;
}

// decoder GRU gate update; H=1024 (y_f/y_bf nullable)
__global__ void k_gru_gates_dec(const float* __restrict__ gi,   // [32][3072] slice
                                const float* __restrict__ gh,   // [32][3072]
                                float* __restrict__ h_f,        // [32][1024]
                                u16*   __restrict__ h_bf,
                                float* __restrict__ y_f,
                                u16*   __restrict__ y_bf) {
  int idx = blockIdx.x * 256 + threadIdx.x;    // 0 .. 32*1024-1
  int b = idx >> 10, j = idx & 1023;
  const float* gir = gi + (long)b * 3072;
  const float* ghr = gh + (long)b * 3072;
  float r = sigm(gir[j] + ghr[j]);
  float z = sigm(gir[1024 + j] + ghr[1024 + j]);
  float n = tanhf(gir[2048 + j] + r * ghr[2048 + j]);
  long hoff = (long)b * 1024 + j;
  float hn = (1.0f - z) * n + z * h_f[hoff];
  h_f[hoff] = hn;  h_bf[hoff] = f2bf(hn);
  if (y_f)  y_f[hoff]  = hn;
  if (y_bf) y_bf[hoff] = f2bf(hn);
}

// save encoder final hidden (cat fwd|bwd) as decoder init hidden for layer l
__global__ void k_save_hidden(const float* __restrict__ h_f,    // [2][32][512]
                              float* __restrict__ dh_f,         // [2][32][1024]
                              u16*   __restrict__ dh_bf, int layer) {
  int idx = blockIdx.x * 256 + threadIdx.x;    // d*32*512 + b*512 + j
  if (idx >= 2 * 32 * 512) return;
  int j = idx & 511, b = (idx >> 9) & 31, d = idx >> 14;
  float v = h_f[idx];
  long o = (long)layer * 32 * 1024 + (long)b * 1024 + (long)d * 512 + j;
  dh_f[o] = v;  dh_bf[o] = f2bf(v);
}

// enc_outputs [t,b,h] f32 -> per-batch transposed bf16 [b][h][t] (W^T operand)
__global__ void k_enc_transpose(const float* __restrict__ y, u16* __restrict__ encT) {
  long idx = (long)blockIdx.x * 256 + threadIdx.x;  // b*1024*128 + h*128 + t
  if (idx >= 32L * 1024 * 128) return;
  int  t = (int)(idx & 127);
  long rest = idx >> 7;
  int  h = (int)(rest & 1023);
  int  b = (int)(rest >> 10);
  encT[idx] = f2bf(y[((long)t * 32 + b) * 1024 + h]);
}

// softmax over L_in=128 per (b, t_out); writes f32 attn to d_out + bf16 copy
__global__ void k_softmax(const float* __restrict__ scores,     // [b][64][128]
                          float* __restrict__ attn_out,         // [t][b][128]
                          u16*   __restrict__ attn_bf) {
  int t = blockIdx.x, b = blockIdx.y, i = threadIdx.x;
  __shared__ float red[128];
  float v = scores[((long)b * 64 + t) * 128 + i];
  red[i] = v; __syncthreads();
  for (int s = 64; s > 0; s >>= 1) { if (i < s) red[i] = fmaxf(red[i], red[i + s]); __syncthreads(); }
  float m = red[0]; __syncthreads();
  float e = __expf(v - m);
  red[i] = e; __syncthreads();
  for (int s = 64; s > 0; s >>= 1) { if (i < s) red[i] += red[i + s]; __syncthreads(); }
  float a = e / red[0];
  long o = ((long)t * 32 + b) * 128 + i;
  attn_out[o] = a;  attn_bf[o] = f2bf(a);
}

// cat(ctx, rnn_out) -> bf16 [t*32+b][2048]
__global__ void k_concat(const float* __restrict__ ctx, const float* __restrict__ rnn,
                         u16* __restrict__ cat) {
  long idx = (long)blockIdx.x * 256 + threadIdx.x;
  if (idx >= 2048L * 2048) return;
  long row = idx >> 11; int j = (int)(idx & 2047);
  float v = (j < 1024) ? ctx[row * 1024 + j] : rnn[row * 1024 + (j - 1024)];
  cat[idx] = f2bf(v);
}

// ---------------------------------------------------------------------------

extern "C" void kernel_launch(void* const* d_in, const int* in_sizes, int n_in,
                              void* d_out, int out_size, void* d_ws, size_t ws_size,
                              hipStream_t stream) {
  (void)in_sizes; (void)n_in; (void)out_size; (void)ws_size;
  const int LIN = 128, LOUT = 64, B = 32, H = 1024, HH = 512, V = 32000;

  const int*   input_tokens  = (const int*)d_in[0];
  const int*   output_tokens = (const int*)d_in[2];
  const float* enc_embedding = (const float*)d_in[4];
  const float* enc_b_ih      = (const float*)d_in[7];
  const float* enc_b_hh      = (const float*)d_in[8];
  const float* dec_embedding = (const float*)d_in[9];
  const float* dec_b_ih      = (const float*)d_in[12];
  const float* dec_b_hh      = (const float*)d_in[13];
  const float* joiner_b      = (const float*)d_in[16];
  const float* proj_b        = (const float*)d_in[18];

  float* out_logits = (float*)d_out;
  float* out_attn   = out_logits + (size_t)LOUT * B * V;

  // ---- workspace layout ----
  char* wsb = (char*)d_ws;  size_t off = 0;
  auto alloc = [&](size_t bytes) -> char* {
    off = (off + 255) & ~(size_t)255;
    char* p = wsb + off; off += bytes; return p;
  };
  u16*   x_bf    = (u16*)  alloc((size_t)LIN * B * H * 2);         // enc layer input bf16
  float* gi_enc  = (float*)alloc((size_t)2 * LIN * B * 1536 * 4);  // [dir][L*B][3HH]
  float* gh_enc  = (float*)alloc((size_t)2 * B * 1536 * 4);
  float* h_f     = (float*)alloc((size_t)2 * B * HH * 4);
  u16*   h_bf    = (u16*)  alloc((size_t)2 * B * HH * 2);
  float* y_enc   = (float*)alloc((size_t)LIN * B * H * 4);         // layer output / enc_outputs
  u16*   wih_bf  = (u16*)  alloc((size_t)2 * 2 * 1536 * H * 2);
  u16*   whh_bf  = (u16*)  alloc((size_t)2 * 2 * 1536 * HH * 2);
  u16*   dwih_bf = (u16*)  alloc((size_t)2 * 3072 * H * 2);
  u16*   dwhh_bf = (u16*)  alloc((size_t)2 * 3072 * H * 2);
  u16*   attnW_bf= (u16*)  alloc((size_t)H * H * 2);
  u16*   jw_bf   = (u16*)  alloc((size_t)H * 2 * H * 2);
  u16*   pw_bf   = (u16*)  alloc((size_t)V * H * 2);
  float* penc_f  = (float*)alloc((size_t)LIN * B * H * 4);         // proj_enc f32
  u16*   penc_bf = (u16*)  alloc((size_t)LIN * B * H * 2);
  u16*   encT_bf = (u16*)  alloc((size_t)B * H * LIN * 2);         // [b][h][t]
  u16*   din_bf  = (u16*)  alloc((size_t)LOUT * B * H * 2);
  float* gi0_dec = (float*)alloc((size_t)LOUT * B * 3072 * 4);
  float* gi1_dec = (float*)alloc((size_t)B * 3072 * 4);
  float* gh_dec  = (float*)alloc((size_t)B * 3072 * 4);
  float* dh_f    = (float*)alloc((size_t)2 * B * H * 4);           // decoder hidden
  u16*   dh_bf   = (u16*)  alloc((size_t)2 * B * H * 2);
  float* rnn_f   = (float*)alloc((size_t)LOUT * B * H * 4);
  u16*   rnn_bf  = (u16*)  alloc((size_t)LOUT * B * H * 2);
  float* scores  = (float*)alloc((size_t)B * LOUT * LIN * 4);      // [b][t_out][t_in]
  u16*   attn_bf = (u16*)  alloc((size_t)LOUT * B * LIN * 2);      // [t][b][t_in]
  float* ctx_f   = (float*)alloc((size_t)LOUT * B * H * 4);
  u16*   cat_bf  = (u16*)  alloc((size_t)LOUT * B * 2 * H * 2);
  float* vecs_f  = (float*)alloc((size_t)LOUT * B * H * 4);
  u16*   vecs_bf = (u16*)  alloc((size_t)LOUT * B * H * 2);

  auto gemm = [&](const u16* A, long lda, long bsA, const u16* W, long ldw, long bsW,
                  float* C, long ldc, long bsC, const float* bias, long bsBias,
                  int M, int N, int K, int batch, int act) {
    dim3 g((unsigned)(N / 64), (unsigned)(M / 32), (unsigned)batch);
    k_gemm<<<g, 32, 0, stream>>>(A, lda, bsA, W, ldw, bsW, C, ldc, bsC, bias, bsBias, K, act);
  };
  auto cvt = [&](const float* in, u16* out, long n) {
    k_cvt<<<(unsigned)((n + 255) / 256), 256, 0, stream>>>(in, out, n);
  };

  // ---- convert weights f32 -> bf16 (once per call) ----
  cvt((const float*)d_in[5],  wih_bf,  (long)2 * 2 * 1536 * H);
  cvt((const float*)d_in[6],  whh_bf,  (long)2 * 2 * 1536 * HH);
  cvt((const float*)d_in[10], dwih_bf, (long)2 * 3072 * H);
  cvt((const float*)d_in[11], dwhh_bf, (long)2 * 3072 * H);
  cvt((const float*)d_in[14], attnW_bf,(long)H * H);
  cvt((const float*)d_in[15], jw_bf,   (long)H * 2 * H);
  cvt((const float*)d_in[17], pw_bf,   (long)V * H);

  // ---- encoder ----
  k_embed_enc<<<LIN * B, 256, 0, stream>>>(input_tokens, enc_embedding, x_bf);
  for (int l = 0; l < 2; ++l) {
    // batched input projection, both directions: gi = x @ W_ih^T + b_ih
    gemm(x_bf, H, 0,
         wih_bf + (size_t)l * 2 * 1536 * H, H, (long)1536 * H,
         gi_enc, 1536, (long)LIN * B * 1536,
         enc_b_ih + (size_t)l * 2 * 1536, 1536,
         LIN * B, 1536, H, 2, 0);
    // h0 = 0
    k_zero<<<(2 * B * HH + 255) / 256, 256, 0, stream>>>((unsigned*)h_f,  2 * B * HH);
    k_zero<<<(2 * B * HH / 2 + 255) / 256, 256, 0, stream>>>((unsigned*)h_bf, 2 * B * HH / 2);
    // sequential recurrence (fwd dir processes t, bwd dir processes 127-t)
    for (int t = 0; t < LIN; ++t) {
      gemm(h_bf, HH, (long)B * HH,
           whh_bf + (size_t)l * 2 * 1536 * HH, HH, (long)1536 * HH,
           gh_enc, 1536, (long)B * 1536,
           enc_b_hh + (size_t)l * 2 * 1536, 1536,
           B, 1536, HH, 2, 0);
      dim3 gg(B * HH / 256, 2);
      k_gru_gates_enc<<<gg, 256, 0, stream>>>(gi_enc, gh_enc, h_f, h_bf, y_enc, t);
    }
    k_save_hidden<<<(2 * B * HH + 255) / 256, 256, 0, stream>>>(h_f, dh_f, dh_bf, l);
    cvt(y_enc, x_bf, (long)LIN * B * H);   // next layer input / final enc_outputs bf16
  }

  // ---- attention pre-projection: proj_enc = enc_outputs @ attn_W^T ----
  gemm(x_bf, H, 0, attnW_bf, H, 0, penc_f, H, 0, nullptr, 0, LIN * B, H, H, 1, 0);
  cvt(penc_f, penc_bf, (long)LIN * B * H);
  k_enc_transpose<<<(unsigned)((32L * 1024 * 128 + 255) / 256), 256, 0, stream>>>(y_enc, encT_bf);

  // ---- decoder ----
  k_embed_dec<<<LOUT * B, 256, 0, stream>>>(output_tokens, dec_embedding, din_bf);
  // batched layer-0 input projection (teacher forcing: inputs known)
  gemm(din_bf, H, 0, dwih_bf, H, 0, gi0_dec, 3072, 0,
       dec_b_ih, 0, LOUT * B, 3072, H, 1, 0);

  for (int t = 0; t < LOUT; ++t) {
    // layer 0
    gemm(dh_bf, H, 0, dwhh_bf, H, 0, gh_dec, 3072, 0,
         dec_b_hh, 0, B, 3072, H, 1, 0);
    k_gru_gates_dec<<<B * H / 256, 256, 0, stream>>>(
        gi0_dec + (size_t)t * B * 3072, gh_dec, dh_f, dh_bf, nullptr, nullptr);
    // layer 1: gi from layer-0 hidden, gh from layer-1 hidden
    gemm(dh_bf, H, 0, dwih_bf + (size_t)3072 * H, H, 0, gi1_dec, 3072, 0,
         dec_b_ih + 3072, 0, B, 3072, H, 1, 0);
    gemm(dh_bf + (size_t)B * H, H, 0, dwhh_bf + (size_t)3072 * H, H, 0, gh_dec, 3072, 0,
         dec_b_hh + 3072, 0, B, 3072, H, 1, 0);
    k_gru_gates_dec<<<B * H / 256, 256, 0, stream>>>(
        gi1_dec, gh_dec, dh_f + (size_t)B * H, dh_bf + (size_t)B * H,
        rnn_f + (size_t)t * B * H, rnn_bf + (size_t)t * B * H);
  }

  // ---- attention (batched over B): scores = rnn_out . proj_enc ----
  gemm(rnn_bf, (long)B * H, H,
       penc_bf, (long)B * H, H,
       scores, LIN, (long)LOUT * LIN,
       nullptr, 0, LOUT, LIN, H, B, 0);
  { dim3 gs(LOUT, B); k_softmax<<<gs, LIN, 0, stream>>>(scores, out_attn, attn_bf); }
  // ctx = attn @ enc_outputs (via per-batch transposed enc_outputs)
  gemm(attn_bf, (long)B * LIN, LIN,
       encT_bf, LIN, (long)H * LIN,
       ctx_f, (long)B * H, H,
       nullptr, 0, LOUT, H, LIN, B, 0);

  // ---- joiner + vocab projection (fully batched over all 64 steps) ----
  k_concat<<<(unsigned)((2048L * 2048 + 255) / 256), 256, 0, stream>>>(ctx_f, rnn_f, cat_bf);
  gemm(cat_bf, 2 * H, 0, jw_bf, 2 * H, 0, vecs_f, H, 0,
       joiner_b, 0, LOUT * B, H, 2 * H, 1, 1 /*tanh*/);
  cvt(vecs_f, vecs_bf, (long)LOUT * B * H);
  gemm(vecs_bf, H, 0, pw_bf, H, 0, out_logits, V, 0,
       proj_b, 0, LOUT * B, V, H, 1, 0);
}
